// SparseDiffAttn_32212254720698
// MI455X (gfx1250) — matrix operand order repaired and checked by reading, not compiled, not run on hardware
//
#include <hip/hip_runtime.h>

#define H_   12
#define N_   2048
#define D_   128
#define QG_  32
#define BM_  64
#define TOPK_ 192
#define WHALF_ 153
#define SCALE_ 0.08838834764831845f   // 1/sqrt(128)

typedef __bf16 bf16x16 __attribute__((ext_vector_type(16)));
typedef float  v8f     __attribute__((ext_vector_type(8)));

union ABu { bf16x16 v; uint4 q[2]; unsigned short u[16]; };

__device__ __forceinline__ unsigned short f2bf(float f) {
  unsigned int x = __float_as_uint(f);
  x += 0x7fffu + ((x >> 16) & 1u);          // round-to-nearest-even
  return (unsigned short)(x >> 16);
}

// ---- DPP16 row-scope butterfly reductions (lanes 0-15 / 16-31 are DPP rows) ----
// Used only ONCE per kernel per row now (not per tile), so cost is negligible.
template<int CTRL>
__device__ __forceinline__ float dppmov(float v) {
  return __int_as_float(
      __builtin_amdgcn_mov_dpp(__float_as_int(v), CTRL, 0xF, 0xF, true));
}
// 0xB1 = quad_perm[1,0,3,2] (xor1), 0x4E = quad_perm[2,3,0,1] (xor2),
// 0x141 = row_half_mirror (combines quads within 8), 0x140 = row_mirror (combines 8s)
__device__ __forceinline__ float halfmax(float v) {
  v = fmaxf(v, dppmov<0xB1>(v));
  v = fmaxf(v, dppmov<0x4E>(v));
  v = fmaxf(v, dppmov<0x141>(v));
  v = fmaxf(v, dppmov<0x140>(v));
  return v;
}
__device__ __forceinline__ float halfsum(float v) {
  v += dppmov<0xB1>(v);
  v += dppmov<0x4E>(v);
  v += dppmov<0x141>(v);
  v += dppmov<0x140>(v);
  return v;
}

// A operand (16x32 bf16, MxK): lane holds 16 contiguous K elems [(lane>>4)*16, +16)
// of row (lane&15).
__device__ __forceinline__ bf16x16 load_a(const unsigned short* rowbase, int lane) {
  ABu a; int o = (lane >> 4) * 16;
  a.q[0] = *(const uint4*)(rowbase + o);
  a.q[1] = *(const uint4*)(rowbase + o + 8);
  return a.v;
}
// B operand (32x16 bf16, KxN): lane holds 16 contiguous K elems of column (lane&15),
// at K offset (lane>>4)*16.
__device__ __forceinline__ bf16x16 load_b16(const unsigned short* p) {
  ABu b; b.q[0] = *(const uint4*)p; b.q[1] = *(const uint4*)(p + 8);
  return b.v;
}

// S tile (16 rows owned by this wave) x (16 keys starting at n0). Q already in A layout.
__device__ __forceinline__ v8f qk_tile(const bf16x16* aq, const unsigned short* kh,
                                       int n0, int lane) {
  const unsigned short* rb = kh + (size_t)(n0 + (lane & 15)) * D_ + (lane >> 4) * 16;
  ABu b0, b1, b2, b3;
  b0.q[0] = *(const uint4*)(rb +  0); b0.q[1] = *(const uint4*)(rb +   8);
  b1.q[0] = *(const uint4*)(rb + 32); b1.q[1] = *(const uint4*)(rb +  40);
  b2.q[0] = *(const uint4*)(rb + 64); b2.q[1] = *(const uint4*)(rb +  72);
  b3.q[0] = *(const uint4*)(rb + 96); b3.q[1] = *(const uint4*)(rb + 104);
  v8f c = {};
  c = __builtin_amdgcn_wmma_f32_16x16x32_bf16(false, aq[0], false, b0.v, (short)0, c, false, false);
  c = __builtin_amdgcn_wmma_f32_16x16x32_bf16(false, aq[1], false, b1.v, (short)0, c, false, false);
  c = __builtin_amdgcn_wmma_f32_16x16x32_bf16(false, aq[2], false, b2.v, (short)0, c, false, false);
  c = __builtin_amdgcn_wmma_f32_16x16x32_bf16(false, aq[3], false, b3.v, (short)0, c, false, false);
  return c;
}

// ---------------- kernel 1: f32 -> bf16 (Q, K row-major; V transposed to [h][d][n]) ----
__global__ void k_convert(const float* __restrict__ q, const float* __restrict__ k,
                          const float* __restrict__ v, unsigned short* __restrict__ qb,
                          unsigned short* __restrict__ kb, unsigned short* __restrict__ vt) {
  size_t idx = (size_t)blockIdx.x * blockDim.x + threadIdx.x;
  qb[idx] = f2bf(q[idx]);
  kb[idx] = f2bf(k[idx]);
  int d = (int)(idx % D_);
  size_t hn = idx / D_;
  int n = (int)(hn % N_);
  int h = (int)(hn / N_);
  vt[((size_t)h * D_ + d) * N_ + n] = f2bf(v[idx]);
}

// ---------------- kernel 2: dense row softmax stats + per-group colsum bs -------------
// 3 WMMA sweeps with purely elementwise per-tile work; cross-lane butterflies only
// once per sweep. Scale folded after the max sweep (scale > 0 is monotone).
__global__ void k_stats_colsum(const unsigned short* __restrict__ qb,
                               const unsigned short* __restrict__ kb,
                               float* __restrict__ bs) {
  __shared__ float lbs[4][N_];                 // per-wave colsum rows (deterministic sum)
  int blk = blockIdx.x, h = blk / QG_, g = blk % QG_;
  int tid = threadIdx.x, w = tid >> 5, lane = tid & 31;
  const unsigned short* qh = qb + (size_t)h * N_ * D_;
  const unsigned short* kh = kb + (size_t)h * N_ * D_;
  int m0 = g * BM_ + w * 16;
  const unsigned short* qrb = qh + (size_t)(m0 + (lane & 15)) * D_;
  bf16x16 aq[4];
#pragma unroll
  for (int kk = 0; kk < 4; ++kk) aq[kk] = load_a(qrb + kk * 32, lane);

  // sweep 1: raw-score row max (elementwise per lane, one butterfly at end)
  float mx[8];
#pragma unroll
  for (int i = 0; i < 8; ++i) mx[i] = -__builtin_inff();
  for (int n0 = 0; n0 < N_; n0 += 16) {
    v8f s = qk_tile(aq, kh, n0, lane);
#pragma unroll
    for (int i = 0; i < 8; ++i) mx[i] = fmaxf(mx[i], s[i]);
  }
  float m[8];
#pragma unroll
  for (int i = 0; i < 8; ++i) m[i] = halfmax(mx[i]) * SCALE_;

  // sweep 2: sumexp with fixed m (per-lane partials, one butterfly at end)
  float lp[8];
#pragma unroll
  for (int i = 0; i < 8; ++i) lp[i] = 0.f;
  for (int n0 = 0; n0 < N_; n0 += 16) {
    v8f s = qk_tile(aq, kh, n0, lane);
#pragma unroll
    for (int i = 0; i < 8; ++i) lp[i] += __expf(fmaf(s[i], SCALE_, -m[i]));
  }
  float invl[8];
#pragma unroll
  for (int i = 0; i < 8; ++i) invl[i] = 1.0f / halfsum(lp[i]);

  // sweep 3: per-group colsum of probabilities
  for (int n0 = 0; n0 < N_; n0 += 16) {
    v8f s = qk_tile(aq, kh, n0, lane);
    float cs = 0.f;
#pragma unroll
    for (int i = 0; i < 8; ++i) cs = fmaf(__expf(fmaf(s[i], SCALE_, -m[i])), invl[i], cs);
    cs += __shfl_xor(cs, 16, 32);              // combine row-halves of same column
    if (lane < 16) lbs[w][n0 + lane] = cs;     // each column written exactly once
  }
  __syncthreads();
  float* bsg = bs + ((size_t)h * QG_ + g) * N_;
  for (int j = tid; j < N_; j += 128)
    bsg[j] = lbs[0][j] + lbs[1][j] + lbs[2][j] + lbs[3][j];
}

// ---------------- kernel 3: radix-select top-192 over bs row + static window ----------
// bs >= 0, so IEEE bit patterns are order-isomorphic to values. 4 x 8-bit MSB-first
// digit histograms narrow to the 192nd-largest value P; select all > P plus the
// lowest-index elements == P (matches jax.lax.top_k stable tie-break).
__global__ void k_topk_mask(const float* __restrict__ bs, unsigned int* __restrict__ mw) {
  __shared__ unsigned int uvals[N_];
  __shared__ unsigned int hist[256];
  __shared__ unsigned int sel[N_ / 32];
  __shared__ unsigned int sprefix, sremain;
  int blk = blockIdx.x, h = blk / QG_, g = blk % QG_, tid = threadIdx.x;
  const float* src = bs + ((size_t)h * QG_ + g) * N_;
  for (int j = tid; j < N_; j += 256) uvals[j] = __float_as_uint(src[j]);
  if (tid < N_ / 32) sel[tid] = 0u;
  if (tid == 0) { sprefix = 0u; sremain = TOPK_; }
  __syncthreads();
#pragma unroll
  for (int d = 24; d >= 0; d -= 8) {
    hist[tid] = 0u;
    __syncthreads();
    unsigned int pref = sprefix;
    unsigned int himask = (d == 24) ? 0u : (0xFFFFFFFFu << (d + 8));
    for (int j = tid; j < N_; j += 256) {
      unsigned int v = uvals[j];
      if ((v & himask) == pref) atomicAdd(&hist[(v >> d) & 0xFFu], 1u);
    }
    __syncthreads();
    if (tid == 0) {
      unsigned int rem = sremain, cum = 0u;
      for (int b = 255; b >= 0; --b) {
        unsigned int hc = hist[b];
        if (cum + hc >= rem) { sremain = rem - cum; sprefix = pref | ((unsigned int)b << d); break; }
        cum += hc;
      }
    }
    __syncthreads();
  }
  unsigned int P = sprefix;
  for (int j = tid; j < N_; j += 256)
    if (uvals[j] > P) atomicOr(&sel[j >> 5], 1u << (j & 31));
  __syncthreads();
  if (tid == 0) {                      // first r ties by ascending index
    unsigned int r = sremain;
    for (int j = 0; j < N_ && r > 0u; ++j)
      if (uvals[j] == P) { sel[j >> 5] |= 1u << (j & 31); --r; }
  }
  __syncthreads();
  if (tid < N_ / 32) {
    int c = g * BM_ + BM_ / 2;
    int lo = c - WHALF_, hi = c + WHALF_;
    unsigned int wv = sel[tid];
#pragma unroll
    for (int b = 0; b < 32; ++b) {
      int j = tid * 32 + b;
      if (j >= lo && j < hi) wv |= 1u << b;
    }
    mw[((size_t)h * QG_ + g) * (N_ / 32) + tid] = wv;
  }
}

// ---------------- kernel 4: masked flash attention, out = out_cache + P@V ------------
// Masked max pre-pass (elementwise) then a single accumulation pass with FIXED m:
// no per-tile accumulator rescale, no per-tile butterflies.
__global__ void k_sparse_attn(const unsigned short* __restrict__ qb,
                              const unsigned short* __restrict__ kb,
                              const unsigned short* __restrict__ vt,
                              const float* __restrict__ oc,
                              const unsigned int* __restrict__ mw,
                              float* __restrict__ out) {
  __shared__ unsigned int smask[N_ / 32];
  __shared__ __align__(16) unsigned short pst[4][16 * 32];  // per-wave P staging (A layout)
  int blk = blockIdx.x, h = blk / QG_, g = blk % QG_;
  int tid = threadIdx.x, w = tid >> 5, lane = tid & 31;
  if (tid < N_ / 32) smask[tid] = mw[((size_t)h * QG_ + g) * (N_ / 32) + tid];
  __syncthreads();
  const unsigned short* qh = qb + (size_t)h * N_ * D_;
  const unsigned short* kh = kb + (size_t)h * N_ * D_;
  const unsigned short* vh = vt + (size_t)h * N_ * D_;   // [d][n] bf16
  int m0 = g * BM_ + w * 16;
  const unsigned short* qrb = qh + (size_t)(m0 + (lane & 15)) * D_;
  bf16x16 aq[4];
#pragma unroll
  for (int kk = 0; kk < 4; ++kk) aq[kk] = load_a(qrb + kk * 32, lane);

  // pass 1: masked raw-score row max (elementwise), one butterfly at end
  float mx[8];
#pragma unroll
  for (int i = 0; i < 8; ++i) mx[i] = -__builtin_inff();
  for (int n0 = 0; n0 < N_; n0 += 16) {
    unsigned int bits = (smask[n0 >> 5] >> ((n0 & 16))) & 0xFFFFu;
    if (bits == 0u) continue;
    v8f s = qk_tile(aq, kh, n0, lane);
    bool on = (bits >> (lane & 15)) & 1u;
#pragma unroll
    for (int i = 0; i < 8; ++i) {
      float x = on ? s[i] : -__builtin_inff();
      mx[i] = fmaxf(mx[i], x);
    }
  }
  float m[8];
#pragma unroll
  for (int i = 0; i < 8; ++i) m[i] = halfmax(mx[i]) * SCALE_;   // finite: mask nonempty

  // pass 2: p = exp(s*scale - m); accumulate l per-lane and O via WMMA
  v8f acc[8] = {};
  float lp[8];
#pragma unroll
  for (int i = 0; i < 8; ++i) lp[i] = 0.f;
  unsigned short* myp = pst[w];
  int arow = lane & 15;

  for (int n0 = 0; n0 < N_; n0 += 32) {
    unsigned int word = smask[n0 >> 5];
    if (word == 0u) continue;                 // uniform skip across block
#pragma unroll
    for (int hh = 0; hh < 2; ++hh) {
      unsigned int bits = (word >> (hh * 16)) & 0xFFFFu;
      if (bits == 0u) {                       // stage zeros for empty half
#pragma unroll
        for (int i = 0; i < 8; ++i) {
          int row = i + ((lane >> 4) << 3);
          myp[row * 32 + hh * 16 + (lane & 15)] = 0;
        }
        continue;
      }
      v8f s = qk_tile(aq, kh, n0 + hh * 16, lane);
      bool on = (bits >> (lane & 15)) & 1u;
#pragma unroll
      for (int i = 0; i < 8; ++i) {
        float t = on ? fmaf(s[i], SCALE_, -m[i]) : -__builtin_inff();
        float p = __expf(t);                  // masked -> 0
        lp[i] += p;
        int row = i + ((lane >> 4) << 3);     // C-layout row for this VGPR slot
        myp[row * 32 + hh * 16 + (lane & 15)] = f2bf(p);
      }
    }
    // same-wave DS in-order: staged writes visible to our own reads
    ABu pa;
    {
      int o = (lane >> 4) * 16;
      const unsigned short* pp = myp + arow * 32;
      pa.q[0] = *(const uint4*)(pp + o);
      pa.q[1] = *(const uint4*)(pp + o + 8);
    }
#pragma unroll
    for (int dc = 0; dc < 8; ++dc) {
      bf16x16 b = load_b16(vh + (size_t)(dc * 16 + (lane & 15)) * N_ +
                           n0 + (lane >> 4) * 16);
      acc[dc] = __builtin_amdgcn_wmma_f32_16x16x32_bf16(false, pa.v, false, b,
                                                        (short)0, acc[dc], false, false);
    }
  }
  float invl[8];
#pragma unroll
  for (int i = 0; i < 8; ++i) invl[i] = 1.0f / halfsum(lp[i]);  // mask nonempty -> l > 0

  // epilogue: out = out_cache + acc / l
#pragma unroll
  for (int dc = 0; dc < 8; ++dc) {
#pragma unroll
    for (int i = 0; i < 8; ++i) {
      int row = m0 + i + ((lane >> 4) << 3);
      int col = dc * 16 + (lane & 15);
      size_t idx = ((size_t)h * N_ + row) * D_ + col;
      out[idx] = fmaf(acc[dc][i], invl[i], oc[idx]);
    }
  }
}

extern "C" void kernel_launch(void* const* d_in, const int* in_sizes, int n_in,
                              void* d_out, int out_size, void* d_ws, size_t ws_size,
                              hipStream_t stream) {
  (void)in_sizes; (void)n_in; (void)out_size; (void)ws_size;
  const float* q  = (const float*)d_in[0];
  const float* k  = (const float*)d_in[1];
  const float* v  = (const float*)d_in[2];
  const float* oc = (const float*)d_in[3];
  float* out = (float*)d_out;

  // workspace layout (~21.1 MB): qb | kb | vt (bf16) | bs (f32) | mask (u32)
  unsigned short* qb = (unsigned short*)d_ws;
  unsigned short* kb = qb + (size_t)H_ * N_ * D_;
  unsigned short* vt = kb + (size_t)H_ * N_ * D_;
  float* bs = (float*)(vt + (size_t)H_ * N_ * D_);
  unsigned int* mw = (unsigned int*)(bs + (size_t)H_ * QG_ * N_);

  k_convert<<<(H_ * N_ * D_) / 256, 256, 0, stream>>>(q, k, v, qb, kb, vt);
  k_stats_colsum<<<H_ * QG_, 128, 0, stream>>>(qb, kb, bs);
  k_topk_mask<<<H_ * QG_, 256, 0, stream>>>(bs, mw);
  k_sparse_attn<<<H_ * QG_, 128, 0, stream>>>(qb, kb, vt, oc, mw, out);
}
